// HybridStateSpaceAttention_9028021256798
// MI455X (gfx1250) — matrix operand order repaired and checked by reading, not compiled, not run on hardware
//
#include <hip/hip_runtime.h>
#include <hip/hip_bf16.h>
#include <math.h>

#define H_DIM 1024
#define NHEAD 16
#define HD    64
#define WIN   512
#define STR   256
#define GMEM  64
#define BATCH 2
#define SEQ   4096
#define NW    15          // (4096-512)/256+1
#define GT2   128         // 2*G

typedef __bf16 bf16;
typedef __attribute__((ext_vector_type(16))) __bf16 v16bf;
typedef __attribute__((ext_vector_type(8)))  float  v8f;

// ---------------------------------------------------------------------------
// WMMA helpers: CDNA5 v_wmma_f32_16x16x32_bf16, D = A(16x32) * B(32x16) + C
// ---------------------------------------------------------------------------
__device__ __forceinline__ v8f wmma_bf16(v16bf a, v16bf b, v8f c) {
  return __builtin_amdgcn_wmma_f32_16x16x32_bf16(
      /*neg_a=*/false, a, /*neg_b=*/false, b,
      /*c_mod=*/(short)0, c, /*reuse_a=*/false, /*reuse_b=*/false);
}

// A-matrix 16x32 bf16 gather (row-major source, leading dim ld).
__device__ __forceinline__ v16bf load_A16x32(const bf16* base, int ld) {
  int lane = threadIdx.x & 31;
  int row  = lane & 15;
  int half = lane >> 4;
  v16bf a;
#pragma unroll
  for (int e = 0; e < 16; ++e) {
    int k = half * 16 + e;
    a[e] = base[row * ld + k];
  }
  return a;
}

// B-matrix 32x16 bf16 where Bmat[k,n] = base[k*ld + n] (row-major K x N tile).
__device__ __forceinline__ v16bf load_B_rowmajor(const bf16* base, int ld) {
  int lane = threadIdx.x & 31;
  int col  = lane & 15;
  int half = lane >> 4;
  v16bf b;
#pragma unroll
  for (int e = 0; e < 16; ++e) b[e] = base[(half * 16 + e) * ld + col];
  return b;
}

// B-matrix 32x16 bf16 where Bmat[k,n] = base[n*ld + k] (used for Q*K^T).
__device__ __forceinline__ v16bf load_B_colmajor(const bf16* base, int ld) {
  int lane = threadIdx.x & 31;
  int col  = lane & 15;
  int half = lane >> 4;
  v16bf b;
#pragma unroll
  for (int e = 0; e < 16; ++e) b[e] = base[col * ld + half * 16 + e];
  return b;
}

__device__ __forceinline__ float redmax16(float v) {
#pragma unroll
  for (int off = 1; off < 16; off <<= 1) v = fmaxf(v, __shfl_xor(v, off, 32));
  return v;
}
__device__ __forceinline__ float redsum16(float v) {
#pragma unroll
  for (int off = 1; off < 16; off <<= 1) v += __shfl_xor(v, off, 32);
  return v;
}

// ---------------------------------------------------------------------------
// CDNA5 async global->LDS staging (ASYNCcnt path).
// VDST = per-lane LDS byte offset, VADDR = 64-bit global address.
// Low 32 bits of a generic __shared__ pointer are the LDS byte address
// (flat->LDS aperture mapping truncates to addr[31:0]).
// ---------------------------------------------------------------------------
__device__ __forceinline__ void async_ld_b128(unsigned ldsOff, const void* g) {
  unsigned long long ga = (unsigned long long)g;
  asm volatile("global_load_async_to_lds_b128 %0, %1, off"
               :: "v"(ldsOff), "v"(ga) : "memory");
}
__device__ __forceinline__ void wait_async0() {
  asm volatile("s_wait_asynccnt 0x0" ::: "memory");
}

// Stage a rows x cols bf16 tile (cols multiple of 8, 16B-aligned src) into LDS.
__device__ __forceinline__ void stage_tile_async(bf16* dstLds, const bf16* src,
                                                 int rows, int cols, int srcLd) {
  unsigned base = (unsigned)(unsigned long long)(const void*)dstLds;
  int cpr = cols >> 3;              // 16B chunks per row
  int chunks = rows * cpr;
  for (int i = threadIdx.x; i < chunks; i += blockDim.x) {
    int r = i / cpr;
    int cc = (i - r * cpr) << 3;
    async_ld_b128(base + (unsigned)i * 16u, src + (size_t)r * srcLd + cc);
  }
}

// ---------------------------------------------------------------------------
// Elementwise / prep kernels
// ---------------------------------------------------------------------------
__global__ void k_f32_to_bf16(const float* __restrict__ in, bf16* __restrict__ out, int n) {
  int i = blockIdx.x * blockDim.x + threadIdx.x;
  if (i < n) out[i] = (bf16)in[i];
}

// comp[b,g,o] = conv_b[o] + sum_{i,k} x[b,4g+k,i] * conv_w[o,i,k]; fills gt rows 0..63
__global__ void k_conv_compress(const float* __restrict__ x, const float* __restrict__ conv_w,
                                const float* __restrict__ conv_b,
                                float* __restrict__ gt, bf16* __restrict__ gt_bf) {
  __shared__ float xs[4096];
  int bg = blockIdx.x;            // b*G + g
  int b = bg >> 6, g = bg & 63;
  for (int j = threadIdx.x; j < 4096; j += blockDim.x) {
    int k = j >> 10, i = j & 1023;
    xs[j] = x[((size_t)(b * SEQ + 4 * g + k)) * H_DIM + i];
  }
  __syncthreads();
  for (int o = threadIdx.x; o < H_DIM; o += blockDim.x) {
    float acc = conv_b[o];
    const float* w = conv_w + (size_t)o * 4096;
#pragma unroll 4
    for (int i = 0; i < 1024; ++i) {
      acc += xs[i] * w[i * 4 + 0] + xs[1024 + i] * w[i * 4 + 1] +
             xs[2048 + i] * w[i * 4 + 2] + xs[3072 + i] * w[i * 4 + 3];
    }
    size_t idx = ((size_t)(b * GT2 + g)) * H_DIM + o;
    gt[idx] = acc;
    gt_bf[idx] = (bf16)acc;
  }
}

// gt rows 64..127: mem[g,o] = gm[0, o/64, g, o%64], broadcast over batch
__global__ void k_fill_gt_mem(const float* __restrict__ gm, float* __restrict__ gt,
                              bf16* __restrict__ gt_bf) {
  int i = blockIdx.x * blockDim.x + threadIdx.x;  // B*G*H
  if (i >= BATCH * GMEM * H_DIM) return;
  int o = i & 1023;
  int g = (i >> 10) & 63;
  int b = i >> 16;
  int h = o >> 6, d = o & 63;
  float v = gm[((size_t)h * GMEM + g) * HD + d];
  size_t idx = ((size_t)(b * GT2 + GMEM + g)) * H_DIM + o;
  gt[idx] = v;
  gt_bf[idx] = (bf16)v;
}

// den[s] = sum of tri weights of windows covering s
__global__ void k_den(float* __restrict__ den) {
  int n = blockIdx.x;
  int w = threadIdx.x;              // blockDim = WIN
  float tri = 0.5f + (float)w * (1.0f / 511.0f);
  atomicAdd(&den[n * STR + w], tri);
}

__global__ void k_finalize_local(const float* __restrict__ num, const float* __restrict__ den,
                                 float* __restrict__ lf, bf16* __restrict__ lbf, int n) {
  int i = blockIdx.x * blockDim.x + threadIdx.x;
  if (i >= n) return;
  int s = (i / H_DIM) % SEQ;
  float v = num[i] / (den[s] + 1e-6f);
  lf[i] = v;
  lbf[i] = (bf16)v;
}

// one roll-and-add diffusion step: cout = cin + 0.5*(cin[s-sh] + cin[s+sh]); res += cout
__global__ void k_bcast_step(const float* __restrict__ cin, float* __restrict__ cout,
                             float* __restrict__ res, int shift, int n) {
  int i = blockIdx.x * blockDim.x + threadIdx.x;
  if (i >= n) return;
  int c = i & (H_DIM - 1);
  int t = i >> 10;
  int s = t & (SEQ - 1);
  int b = t >> 12;
  int sm = s - shift; if (sm < 0) sm += SEQ;
  int sp = s + shift; if (sp >= SEQ) sp -= SEQ;
  float v = cin[i] + 0.5f * (cin[((size_t)(b * SEQ + sm)) * H_DIM + c] +
                             cin[((size_t)(b * SEQ + sp)) * H_DIM + c]);
  cout[i] = v;
  res[i] += v;
}

// ---------------------------------------------------------------------------
// Sliding-window self-attention (flash-style online softmax) + triangular fold
// grid.x = B*NW*NH*8 (8 q-tiles of 64 rows); block = 128 (4 waves, 16 q-rows/wave)
// ---------------------------------------------------------------------------
__global__ void __launch_bounds__(128)
k_sliding_attn(const bf16* __restrict__ xbf, float* __restrict__ num) {
  int id = blockIdx.x;
  int qt = id & 7;  id >>= 3;
  int h  = id % NHEAD; id /= NHEAD;
  int n  = id % NW; id /= NW;
  int b  = id;
  int wave = threadIdx.x >> 5;
  int lane = threadIdx.x & 31;
  int half = lane >> 4;
  int lc   = lane & 15;
  int winBase = n * STR;

  __shared__ bf16 Qs[64][64];      // 8 KB
  __shared__ bf16 Kv[128][64];     // 16 KB  (K == V == window slice of x)
  __shared__ bf16 P[4][16][128];   // 16 KB  per-wave probabilities

  stage_tile_async(&Qs[0][0],
                   xbf + ((size_t)(b * SEQ + winBase + qt * 64)) * H_DIM + h * HD,
                   64, 64, H_DIM);
  wait_async0();
  __syncthreads();

  v8f o[4] = {};
  float m[8], l[8];
#pragma unroll
  for (int e = 0; e < 8; ++e) { m[e] = -1e30f; l[e] = 0.0f; }

  v16bf aq0 = load_A16x32(&Qs[wave * 16][0],  64);
  v16bf aq1 = load_A16x32(&Qs[wave * 16][32], 64);

  for (int ch = 0; ch < 4; ++ch) {
    stage_tile_async(&Kv[0][0],
                     xbf + ((size_t)(b * SEQ + winBase + ch * 128)) * H_DIM + h * HD,
                     128, 64, H_DIM);
    wait_async0();
    __syncthreads();

    // S-tile: 16 x 128, 8 column blocks, K-dim = HD = 64 (2 WMMAs each)
    v8f s[8];
#pragma unroll
    for (int cb = 0; cb < 8; ++cb) {
      v8f acc = {};
      acc = wmma_bf16(aq0, load_B_colmajor(&Kv[cb * 16][0],  64), acc);
      acc = wmma_bf16(aq1, load_B_colmajor(&Kv[cb * 16][32], 64), acc);
#pragma unroll
      for (int e = 0; e < 8; ++e) acc[e] *= 0.125f;   // 1/sqrt(HD)
      s[cb] = acc;
    }

    // online softmax update per row (row = e + 8*half, shared by 16 lanes)
#pragma unroll
    for (int e = 0; e < 8; ++e) {
      float mloc = -1e30f;
#pragma unroll
      for (int cb = 0; cb < 8; ++cb) mloc = fmaxf(mloc, s[cb][e]);
      mloc = redmax16(mloc);
      float mnew = fmaxf(m[e], mloc);
      float corr = __expf(m[e] - mnew);
      float lsum = 0.0f;
#pragma unroll
      for (int cb = 0; cb < 8; ++cb) {
        float p = __expf(s[cb][e] - mnew);
        s[cb][e] = p;
        lsum += p;
      }
      lsum = redsum16(lsum);
      l[e] = l[e] * corr + lsum;
      m[e] = mnew;
#pragma unroll
      for (int nb = 0; nb < 4; ++nb) o[nb][e] *= corr;
    }

    // stash probs (own-wave LDS; DS ops are in-order within a wave)
#pragma unroll
    for (int cb = 0; cb < 8; ++cb)
#pragma unroll
      for (int e = 0; e < 8; ++e)
        P[wave][e + 8 * half][cb * 16 + lc] = (bf16)s[cb][e];

    // O += P * V   (V rows = Kv chunk rows)
#pragma unroll
    for (int nb = 0; nb < 4; ++nb)
#pragma unroll
      for (int kb = 0; kb < 4; ++kb)
        o[nb] = wmma_bf16(load_A16x32(&P[wave][0][kb * 32], 128),
                          load_B_rowmajor(&Kv[kb * 32][nb * 16], 64), o[nb]);
    __syncthreads();
  }

  // normalize, apply tri weight, fold into num (overlapping windows -> atomics)
#pragma unroll
  for (int e = 0; e < 8; ++e) {
    int r = e + 8 * half;
    int q = qt * 64 + wave * 16 + r;             // position within window
    int tok = winBase + q;
    float tri = 0.5f + (float)q * (1.0f / 511.0f);
    float inv = tri / l[e];
#pragma unroll
    for (int nb = 0; nb < 4; ++nb) {
      int col = h * HD + nb * 16 + lc;
      atomicAdd(&num[((size_t)(b * SEQ + tok)) * H_DIM + col], o[nb][e] * inv);
    }
  }
}

// ---------------------------------------------------------------------------
// Compressed-global cross-attention: Q [S,64] x gt-head [128,64]
// grid.x = B*NH*(S/64); block = 128 (4 waves)
// ---------------------------------------------------------------------------
__global__ void __launch_bounds__(128)
k_global_attn(const bf16* __restrict__ xbf, const bf16* __restrict__ gtbf,
              float* __restrict__ globf, bf16* __restrict__ globbf) {
  int id = blockIdx.x;
  int qt = id & 63; id >>= 6;
  int h  = id % NHEAD;
  int b  = id / NHEAD;
  int wave = threadIdx.x >> 5;
  int lane = threadIdx.x & 31;
  int half = lane >> 4;
  int lc   = lane & 15;

  __shared__ bf16 Qs[64][64];
  __shared__ bf16 Gs[128][64];
  __shared__ bf16 P[4][16][128];

  stage_tile_async(&Qs[0][0],
                   xbf + ((size_t)(b * SEQ + qt * 64)) * H_DIM + h * HD,
                   64, 64, H_DIM);
  stage_tile_async(&Gs[0][0],
                   gtbf + ((size_t)(b * GT2)) * H_DIM + h * HD,
                   128, 64, H_DIM);
  wait_async0();
  __syncthreads();

  v16bf aq0 = load_A16x32(&Qs[wave * 16][0],  64);
  v16bf aq1 = load_A16x32(&Qs[wave * 16][32], 64);

  v8f s[8];
#pragma unroll
  for (int cb = 0; cb < 8; ++cb) {
    v8f acc = {};
    acc = wmma_bf16(aq0, load_B_colmajor(&Gs[cb * 16][0],  64), acc);
    acc = wmma_bf16(aq1, load_B_colmajor(&Gs[cb * 16][32], 64), acc);
#pragma unroll
    for (int e = 0; e < 8; ++e) acc[e] *= 0.125f;
    s[cb] = acc;
  }

  float l[8];
#pragma unroll
  for (int e = 0; e < 8; ++e) {
    float mloc = -1e30f;
#pragma unroll
    for (int cb = 0; cb < 8; ++cb) mloc = fmaxf(mloc, s[cb][e]);
    mloc = redmax16(mloc);
    float lsum = 0.0f;
#pragma unroll
    for (int cb = 0; cb < 8; ++cb) {
      float p = __expf(s[cb][e] - mloc);
      s[cb][e] = p;
      lsum += p;
    }
    l[e] = redsum16(lsum);
#pragma unroll
    for (int cb = 0; cb < 8; ++cb)
      P[wave][e + 8 * half][cb * 16 + lc] = (bf16)s[cb][e];
  }

  v8f o[4] = {};
#pragma unroll
  for (int nb = 0; nb < 4; ++nb)
#pragma unroll
    for (int kb = 0; kb < 4; ++kb)
      o[nb] = wmma_bf16(load_A16x32(&P[wave][0][kb * 32], 128),
                        load_B_rowmajor(&Gs[kb * 32][nb * 16], 64), o[nb]);

#pragma unroll
  for (int e = 0; e < 8; ++e) {
    int tok = qt * 64 + wave * 16 + e + 8 * half;
    float inv = 1.0f / l[e];
#pragma unroll
    for (int nb = 0; nb < 4; ++nb) {
      size_t idx = ((size_t)(b * SEQ + tok)) * H_DIM + h * HD + nb * 16 + lc;
      float v = o[nb][e] * inv;
      globf[idx] = v;
      globbf[idx] = (bf16)v;
    }
  }
}

// ---------------------------------------------------------------------------
// Gate GEMM: sigmoid([local|glob] @ mix_w + mix_b) fused with the mix epilogue.
// grid (M/64, N/64); block 128 (4 waves, wave = 16 rows x 64 cols)
// ---------------------------------------------------------------------------
__global__ void __launch_bounds__(128)
k_gate_mix(const bf16* __restrict__ lbf, const bf16* __restrict__ gbf,
           const bf16* __restrict__ wbf, const float* __restrict__ bias,
           const float* __restrict__ lf, const float* __restrict__ gf,
           const float* __restrict__ res, bf16* __restrict__ mixedbf) {
  int m0 = blockIdx.x * 64;
  int n0 = blockIdx.y * 64;
  int wave = threadIdx.x >> 5;
  int lane = threadIdx.x & 31;
  int half = lane >> 4;
  int lc   = lane & 15;

  __shared__ bf16 As[64][32];
  __shared__ bf16 Bs[32][64];

  v8f acc[4] = {};
  for (int k0 = 0; k0 < 2 * H_DIM; k0 += 32) {
    // each 32-wide K tile comes wholly from local (k0<1024) or glob
    const bf16* asrc = (k0 < H_DIM) ? lbf : gbf;
    int kofs = (k0 < H_DIM) ? k0 : (k0 - H_DIM);
    stage_tile_async(&As[0][0], asrc + (size_t)m0 * H_DIM + kofs, 64, 32, H_DIM);
    stage_tile_async(&Bs[0][0], wbf + (size_t)k0 * H_DIM + n0, 32, 64, H_DIM);
    wait_async0();
    __syncthreads();
    v16bf a = load_A16x32(&As[wave * 16][0], 32);
#pragma unroll
    for (int nb = 0; nb < 4; ++nb)
      acc[nb] = wmma_bf16(a, load_B_rowmajor(&Bs[0][nb * 16], 64), acc[nb]);
    __syncthreads();
  }

#pragma unroll
  for (int nb = 0; nb < 4; ++nb)
#pragma unroll
    for (int e = 0; e < 8; ++e) {
      int row = m0 + wave * 16 + e + 8 * half;
      int col = n0 + nb * 16 + lc;
      size_t idx = (size_t)row * H_DIM + col;
      float g = 1.0f / (1.0f + __expf(-(acc[nb][e] + bias[col])));
      float v = g * lf[idx] + (1.0f - g) * gf[idx] + res[idx] * (1.0f / 13.0f);
      mixedbf[idx] = (bf16)v;
    }
}

// ---------------------------------------------------------------------------
// Output GEMM: d_out = mixed @ out_w + out_b
// ---------------------------------------------------------------------------
__global__ void __launch_bounds__(128)
k_out_gemm(const bf16* __restrict__ mbf, const bf16* __restrict__ wbf,
           const float* __restrict__ bias, float* __restrict__ out) {
  int m0 = blockIdx.x * 64;
  int n0 = blockIdx.y * 64;
  int wave = threadIdx.x >> 5;
  int lane = threadIdx.x & 31;
  int half = lane >> 4;
  int lc   = lane & 15;

  __shared__ bf16 As[64][32];
  __shared__ bf16 Bs[32][64];

  v8f acc[4] = {};
  for (int k0 = 0; k0 < H_DIM; k0 += 32) {
    stage_tile_async(&As[0][0], mbf + (size_t)m0 * H_DIM + k0, 64, 32, H_DIM);
    stage_tile_async(&Bs[0][0], wbf + (size_t)k0 * H_DIM + n0, 32, 64, H_DIM);
    wait_async0();
    __syncthreads();
    v16bf a = load_A16x32(&As[wave * 16][0], 32);
#pragma unroll
    for (int nb = 0; nb < 4; ++nb)
      acc[nb] = wmma_bf16(a, load_B_rowmajor(&Bs[0][nb * 16], 64), acc[nb]);
    __syncthreads();
  }

#pragma unroll
  for (int nb = 0; nb < 4; ++nb)
#pragma unroll
    for (int e = 0; e < 8; ++e) {
      int row = m0 + wave * 16 + e + 8 * half;
      int col = n0 + nb * 16 + lc;
      out[(size_t)row * H_DIM + col] = acc[nb][e] + bias[col];
    }
}

// ---------------------------------------------------------------------------
extern "C" void kernel_launch(void* const* d_in, const int* in_sizes, int n_in,
                              void* d_out, int out_size, void* d_ws, size_t ws_size,
                              hipStream_t stream) {
  (void)in_sizes; (void)n_in; (void)out_size; (void)ws_size;
  const float* x      = (const float*)d_in[0];
  const float* gm     = (const float*)d_in[1];
  const float* conv_w = (const float*)d_in[2];
  const float* conv_b = (const float*)d_in[3];
  const float* mix_w  = (const float*)d_in[4];
  const float* mix_b  = (const float*)d_in[5];
  const float* out_w  = (const float*)d_in[6];
  const float* out_b  = (const float*)d_in[7];
  float* out = (float*)d_out;

  const size_t N = (size_t)BATCH * SEQ * H_DIM;  // 8,388,608

  char* ws = (char*)d_ws;
  size_t off = 0;
  auto alloc = [&](size_t bytes) -> void* {
    void* p = ws + off;
    off = (off + bytes + 255) & ~(size_t)255;
    return p;
  };

  bf16*  xbf     = (bf16*)alloc(N * sizeof(bf16));
  bf16*  localbf = (bf16*)alloc(N * sizeof(bf16));
  bf16*  globbf  = (bf16*)alloc(N * sizeof(bf16));
  bf16*  mixedbf = (bf16*)alloc(N * sizeof(bf16));
  bf16*  mixwbf  = (bf16*)alloc((size_t)2 * H_DIM * H_DIM * sizeof(bf16));
  bf16*  outwbf  = (bf16*)alloc((size_t)H_DIM * H_DIM * sizeof(bf16));
  float* gt      = (float*)alloc((size_t)BATCH * GT2 * H_DIM * sizeof(float));
  bf16*  gtbf    = (bf16*)alloc((size_t)BATCH * GT2 * H_DIM * sizeof(bf16));
  float* num     = (float*)alloc(N * sizeof(float));
  float* den     = (float*)alloc((size_t)SEQ * sizeof(float));
  float* localf  = (float*)alloc(N * sizeof(float));
  float* globf   = (float*)alloc(N * sizeof(float));
  float* res     = (float*)alloc(N * sizeof(float));
  float* curA    = (float*)alloc(N * sizeof(float));
  float* curB    = (float*)alloc(N * sizeof(float));

  hipMemsetAsync(num, 0, N * sizeof(float), stream);
  hipMemsetAsync(den, 0, (size_t)SEQ * sizeof(float), stream);
  hipMemsetAsync(res, 0, N * sizeof(float), stream);

  const int T = 256;
  // precision conversions
  k_f32_to_bf16<<<(int)((N + T - 1) / T), T, 0, stream>>>(x, xbf, (int)N);
  k_f32_to_bf16<<<(2 * H_DIM * H_DIM + T - 1) / T, T, 0, stream>>>(mix_w, mixwbf, 2 * H_DIM * H_DIM);
  k_f32_to_bf16<<<(H_DIM * H_DIM + T - 1) / T, T, 0, stream>>>(out_w, outwbf, H_DIM * H_DIM);

  // compressed global tokens
  k_conv_compress<<<BATCH * GMEM, 256, 0, stream>>>(x, conv_w, conv_b, gt, gtbf);
  k_fill_gt_mem<<<(BATCH * GMEM * H_DIM + T - 1) / T, T, 0, stream>>>(gm, gt, gtbf);

  // sliding-window attention + triangular fold
  k_den<<<NW, WIN, 0, stream>>>(den);
  k_sliding_attn<<<BATCH * NW * NHEAD * 8, 128, 0, stream>>>(xbf, num);
  k_finalize_local<<<(int)((N + T - 1) / T), T, 0, stream>>>(num, den, localf, localbf, (int)N);

  // O(log n) roll-and-add diffusion (12 steps, ping-pong)
  {
    const float* cin = x;
    float* cout = curA;
    float* other = curB;
    for (int shift = 1; shift < SEQ; shift <<= 1) {
      k_bcast_step<<<(int)((N + T - 1) / T), T, 0, stream>>>(cin, cout, res, shift, (int)N);
      cin = cout;
      cout = other;
      other = (float*)cin;
    }
  }

  // global cross-attention
  k_global_attn<<<BATCH * NHEAD * (SEQ / 64), 128, 0, stream>>>(xbf, gtbf, globf, globbf);

  // gate GEMM + mix epilogue, then output GEMM
  dim3 gGate(BATCH * SEQ / 64, H_DIM / 64);
  k_gate_mix<<<gGate, 128, 0, stream>>>(localbf, globbf, mixwbf, mix_b,
                                        localf, globf, res, mixedbf);
  dim3 gOut(BATCH * SEQ / 64, H_DIM / 64);
  k_out_gemm<<<gOut, 128, 0, stream>>>(mixedbf, outwbf, out_b, out);
}
